// TrustworthyMS_12017318494595
// MI455X (gfx1250) — matrix operand order repaired and checked by reading, not compiled, not run on hardware
//
#include <hip/hip_runtime.h>
#include <hip/hip_bf16.h>
#include <stdint.h>

// ---------------------------------------------------------------------------
// GIN 2-branch model for MI455X (gfx1250, wave32, WMMA).
//  - scatter/gather (bandwidth bound, ~3.6GB @ 23.3 TB/s) via fp32 atomics (L2-resident)
//  - GEMMs via v_wmma_f32_16x16x32_f16, f32 accumulate; A pre-packed to padded f16
//    (fused x+agg), W pre-transposed/padded f16 -> all fragment loads are
//    unconditional aligned vector loads, EXEC all-ones through the WMMA loop
//  - 16x64 output tile per wave (4 WMMAs/k-step, 4x A reuse)
//  - conv2 GEMM fused with mean/max graph pooling (atomicAdd + uint atomicMax on
//    non-negative post-ReLU values) -> avoids a 372MB intermediate
// ---------------------------------------------------------------------------

#define GRP 512
#define EPW 4   // edges per warp in scatter kernel

typedef __attribute__((ext_vector_type(16))) _Float16 v16h;
typedef __attribute__((ext_vector_type(8)))  _Float16 v8h;
typedef __attribute__((ext_vector_type(8)))  float    v8f;

static __device__ __forceinline__ void atomicMaxNonneg(float* addr, float v) {
  // valid ordering for IEEE floats >= 0 (post-ReLU) compared as unsigned ints
  atomicMax((unsigned int*)addr, __float_as_uint(v));
}

__global__ void zero_f32(float* __restrict__ p, long n) {
  long i = (long)blockIdx.x * blockDim.x + threadIdx.x;
  if (i < n) p[i] = 0.0f;
}

__global__ void count_nodes(const int* __restrict__ batch, float* __restrict__ cnt, int n) {
  int i = blockIdx.x * blockDim.x + threadIdx.x;
  if (i < n) atomicAdd(&cnt[batch[i]], 1.0f);
}

// warp-per-edge scatter-add: agg[dst] += x[src]; lanes stride the feature dim.
__global__ void scatter_add(const float* __restrict__ x, const int* __restrict__ ei,
                            float* __restrict__ agg, int E, int F) {
  int warp = (blockIdx.x * blockDim.x + threadIdx.x) >> 5;
  int lane = threadIdx.x & 31;
  long base = (long)warp * EPW;
  for (int i = 0; i < EPW; ++i) {
    long e = base + i;
    if (e >= E) return;
    int s = ei[e];
    int d = ei[(long)E + e];
    if (lane == 0 && i + 1 < EPW && e + 1 < E)
      __builtin_prefetch(x + (long)ei[e + 1] * F, 0, 1);  // global_prefetch_b8
    const float* xs = x + (long)s * F;
    float*       ad = agg + (long)d * F;
    for (int f = lane; f < F; f += 32) atomicAdd(&ad[f], xs[f]);
  }
}

// W [Ka x Kout] f32 row-major -> Wt [Kout_pad x Ka_pad] f16 (transposed, zero-padded)
__global__ void prep_wt(const float* __restrict__ W, _Float16* __restrict__ Wt,
                        int Ka, int Kout, int Ka_pad, int Kout_pad) {
  long i = (long)blockIdx.x * blockDim.x + threadIdx.x;
  long total = (long)Ka_pad * Kout_pad;
  if (i >= total) return;
  int r = (int)(i / Ka_pad);   // output-feature index
  int c = (int)(i % Ka_pad);   // input-feature index
  float v = (r < Kout && c < Ka) ? W[(long)c * Kout + r] : 0.0f;
  Wt[i] = (_Float16)v;
}

// A [M x Ka] f32 (+ optional elementwise A2) -> Apad [M x Ka_pad] f16, zero-padded
__global__ void prep_a(const float* __restrict__ A, const float* __restrict__ A2,
                       _Float16* __restrict__ Apad, int Ka, int Ka_pad, long total) {
  long i = (long)blockIdx.x * blockDim.x + threadIdx.x;
  if (i >= total) return;
  int r = (int)(i / Ka_pad);
  int c = (int)(i % Ka_pad);
  float v = 0.0f;
  if (c < Ka) { v = A[(long)r * Ka + c]; if (A2) v += A2[(long)r * Ka + c]; }
  Apad[i] = (_Float16)v;
}

// One wave computes a 16(m) x 64(n) output tile of  out = op(Apad @ Wt^T + bias).
// flags: bit0 = relu ; bit1 = pooled output (atomic mean/max by batch[]).
// A-fragment lane map (16-bit A 16x32, ISA 7.12.2): lane L holds row M=L%16,
//   elems 0..7  -> K = k + (L/16)*8 + j ; elems 8..15 -> K = k + 16 + (L/16)*8 + (j-8)
// B-fragment: lane L holds col N=L%16, same K map as A (two aligned 16B vector
//   loads per n-tile from the pre-transposed Wt).
__global__ void gemm_wmma(const _Float16* __restrict__ Apad, const _Float16* __restrict__ Wt,
                          const float* __restrict__ bias,
                          float* __restrict__ out,
                          const int* __restrict__ batch,
                          float* __restrict__ pool_sum, float* __restrict__ pool_max,
                          int Ka_pad, int Kout, int flags) {
  const int m0   = blockIdx.x * 16;
  const int n0   = blockIdx.y * 64;
  const int lane = threadIdx.x & 31;
  const int half = lane >> 4;
  const int l15  = lane & 15;

  const _Float16* arow = Apad + (long)(m0 + l15) * Ka_pad + half * 8;
  const _Float16* wrow = Wt + (long)(n0 + l15) * Ka_pad + half * 8;
  const long wstep = (long)16 * Ka_pad;   // next 16-column tile of Wt

  v8f acc[4] = {v8f{}, v8f{}, v8f{}, v8f{}};
  for (int k = 0; k < Ka_pad; k += 32) {
    v8h a0 = *(const v8h*)(arow + k);        // K = k + half*8 + 0..7
    v8h a1 = *(const v8h*)(arow + k + 16);   // K = k + 16 + half*8 + 0..7
    v16h a;
#pragma unroll
    for (int j = 0; j < 8; ++j) { a[j] = a0[j]; a[8 + j] = a1[j]; }
#pragma unroll
    for (int t = 0; t < 4; ++t) {
      v8h b0 = *(const v8h*)(wrow + (long)t * wstep + k);
      v8h b1 = *(const v8h*)(wrow + (long)t * wstep + k + 16);
      v16h b;
#pragma unroll
      for (int j = 0; j < 8; ++j) { b[j] = b0[j]; b[8 + j] = b1[j]; }
      acc[t] = __builtin_amdgcn_wmma_f32_16x16x32_f16(false, a, false, b,
                                                      (short)0, acc[t], false, false);
    }
  }

#pragma unroll
  for (int t = 0; t < 4; ++t) {
    const int c = n0 + t * 16 + l15;
    if (c >= Kout) continue;                // masking only at the store stage
    const float bv = bias ? bias[c] : 0.0f;
#pragma unroll
    for (int v = 0; v < 8; ++v) {
      int r = m0 + v + half * 8;            // C/D lane map (ISA 7.12.2)
      float val = acc[t][v] + bv;
      if (flags & 1) val = fmaxf(val, 0.0f);
      if (flags & 2) {
        int g = batch[r];
        atomicAdd(&pool_sum[(long)g * Kout + c], val);
        atomicMaxNonneg(&pool_max[(long)g * Kout + c], val);
      } else {
        out[(long)r * Kout + c] = val;
      }
    }
  }
}

__global__ void pool_fin(const float* __restrict__ psum, const float* __restrict__ pmax,
                         const float* __restrict__ cnt, float* __restrict__ pooled, int K) {
  long i = (long)blockIdx.x * blockDim.x + threadIdx.x;
  if (i >= (long)GRP * K) return;
  int g = (int)(i / K), c = (int)(i % K);
  float n = fmaxf(cnt[g], 1.0f);
  pooled[(long)g * 2 * K + c]     = psum[i] / n;
  pooled[(long)g * 2 * K + K + c] = pmax[i];
}

// one block per column: population mean/var over rows, then affine normalize
__global__ void batchnorm_col(const float* __restrict__ x, const float* __restrict__ g,
                              const float* __restrict__ b, float* __restrict__ out,
                              int rows, int cols) {
  const int c = blockIdx.x;
  __shared__ float ssum[256], ssq[256];
  float s = 0.f, sq = 0.f;
  for (int r = threadIdx.x; r < rows; r += blockDim.x) {
    float v = x[(long)r * cols + c];
    s += v; sq += v * v;
  }
  ssum[threadIdx.x] = s; ssq[threadIdx.x] = sq;
  __syncthreads();
  for (int off = blockDim.x >> 1; off > 0; off >>= 1) {
    if ((int)threadIdx.x < off) {
      ssum[threadIdx.x] += ssum[threadIdx.x + off];
      ssq[threadIdx.x]  += ssq[threadIdx.x + off];
    }
    __syncthreads();
  }
  float mean  = ssum[0] / rows;
  float var   = ssq[0] / rows - mean * mean;
  float scale = g[c] * rsqrtf(var + 1e-5f);
  float shift = b[c] - scale * mean;
  for (int r = threadIdx.x; r < rows; r += blockDim.x)
    out[(long)r * cols + c] = scale * x[(long)r * cols + c] + shift;
}

// ---------------------------------------------------------------------------

static inline int pad32i(int x) { return (x + 31) & ~31; }
static inline int pad64i(int x) { return (x + 63) & ~63; }

extern "C" void kernel_launch(void* const* d_in, const int* in_sizes, int n_in,
                              void* d_out, int out_size, void* d_ws, size_t ws_size,
                              hipStream_t stream) {
  (void)in_sizes; (void)n_in; (void)out_size; (void)ws_size;

  const int N = 100000, E = 1600000, F0 = 93, F1 = 43;
  const int K20 = F0 * 10, K21 = F1 * 10;

  const float* x0  = (const float*)d_in[0];
  const float* x1  = (const float*)d_in[1];
  const int*   ei0 = (const int*)d_in[2];
  const int*   ei1 = (const int*)d_in[3];
  const int*   bt0 = (const int*)d_in[4];
  const int*   bt1 = (const int*)d_in[5];

  // workspace bump allocator (~110 MB total)
  char* wsp = (char*)d_ws;
  auto alloc = [&](size_t bytes) -> char* {
    char* p = wsp; wsp += (bytes + 255) & ~(size_t)255; return p;
  };
  float*    agg    = (float*)alloc((size_t)N * F0 * 4);
  float*    h1     = (float*)alloc((size_t)N * F0 * 4);
  _Float16* wt     = (_Float16*)alloc((size_t)1024 * 1888 * 2);    // largest Wt (pad64 rows)
  _Float16* apad   = (_Float16*)alloc((size_t)N * 96 * 2);          // largest Apad
  float*    psum   = (float*)alloc((size_t)GRP * K20 * 4);
  float*    pmaxb  = (float*)alloc((size_t)GRP * K20 * 4);
  float*    cnt    = (float*)alloc((size_t)GRP * 4);
  float*    pooled = (float*)alloc((size_t)GRP * 2 * K20 * 4);
  float*    mlp1   = (float*)alloc((size_t)GRP * 1024 * 4);
  float*    mlp2   = (float*)alloc((size_t)GRP * 512 * 4);
  float*    head1  = (float*)alloc((size_t)GRP * 256 * 4);

  float* out = (float*)d_out;
  float* z0  = out;                        // [512,2]
  float* xg0 = out + 1024;                 // [512,512]
  float* xg1 = xg0 + 512 * 512;            // [512,512]
  float* z1  = xg1 + 512 * 512;            // [512,2]

  auto zero = [&](float* p, long n) {
    zero_f32<<<dim3((unsigned)((n + 255) / 256)), 256, 0, stream>>>(p, n);
  };
  auto gemm = [&](const float* A, const float* A2, const float* W, const float* bias,
                  float* o, const int* batch, float* ps, float* pm,
                  int M, int Ka, int Kout, int flags) {
    int kap = pad32i(Ka), kop = pad64i(Kout);
    long wtot = (long)kap * kop;
    prep_wt<<<dim3((unsigned)((wtot + 255) / 256)), 256, 0, stream>>>(W, wt, Ka, Kout, kap, kop);
    long atot = (long)M * kap;
    prep_a<<<dim3((unsigned)((atot + 255) / 256)), 256, 0, stream>>>(A, A2, apad, Ka, kap, atot);
    gemm_wmma<<<dim3(M / 16, kop / 64), 32, 0, stream>>>(apad, wt, bias, o, batch, ps, pm,
                                                         kap, Kout, flags);
  };

  const int warps   = (E + EPW - 1) / EPW;
  const int sblocks = (warps * 32 + 255) / 256;

  auto run_branch = [&](const float* x, const int* ei, const int* batch,
                        const float* wc1, const float* bc1,
                        const float* wc2, const float* bc2,
                        const float* gw1, const float* gb1,
                        const float* gw2, const float* gb2,
                        const float* bng, const float* bnb,
                        const float* fw1, const float* fb1,
                        const float* fw2, const float* fb2,
                        int F, int K2, float* xg, float* z) {
    // conv1: h1 = relu((x + agg) @ wc1 + bc1)
    zero(agg, (long)N * F);
    scatter_add<<<sblocks, 256, 0, stream>>>(x, ei, agg, E, F);
    gemm(x, agg, wc1, bc1, h1, nullptr, nullptr, nullptr, N, F, F, 1);
    // conv2 fused with mean/max pooling
    zero(agg, (long)N * F);
    scatter_add<<<sblocks, 256, 0, stream>>>(h1, ei, agg, E, F);
    zero(psum, (long)GRP * K2);
    zero(pmaxb, (long)GRP * K2);
    zero(cnt, GRP);
    count_nodes<<<(N + 255) / 256, 256, 0, stream>>>(batch, cnt, N);
    gemm(h1, agg, wc2, bc2, nullptr, batch, psum, pmaxb, N, F, K2, 1 | 2);
    pool_fin<<<(unsigned)(((long)GRP * K2 + 255) / 256), 256, 0, stream>>>(psum, pmaxb, cnt,
                                                                           pooled, K2);
    // graph MLP + batchnorm -> xg (written straight into d_out)
    gemm(pooled, nullptr, gw1, gb1, mlp1, nullptr, nullptr, nullptr, GRP, 2 * K2, 1024, 1);
    gemm(mlp1,   nullptr, gw2, gb2, mlp2, nullptr, nullptr, nullptr, GRP, 1024, 512, 0);
    batchnorm_col<<<512, 256, 0, stream>>>(mlp2, bng, bnb, xg, GRP, 512);
    // head -> z
    gemm(xg,    nullptr, fw1, fb1, head1, nullptr, nullptr, nullptr, GRP, 512, 256, 1);
    gemm(head1, nullptr, fw2, fb2, z,     nullptr, nullptr, nullptr, GRP, 256, 2, 0);
  };

  run_branch(x0, ei0, bt0,
             (const float*)d_in[6],  (const float*)d_in[7],
             (const float*)d_in[8],  (const float*)d_in[9],
             (const float*)d_in[14], (const float*)d_in[15],
             (const float*)d_in[16], (const float*)d_in[17],
             (const float*)d_in[18], (const float*)d_in[19],
             (const float*)d_in[26], (const float*)d_in[27],
             (const float*)d_in[28], (const float*)d_in[29],
             F0, K20, xg0, z0);

  run_branch(x1, ei1, bt1,
             (const float*)d_in[10], (const float*)d_in[11],
             (const float*)d_in[12], (const float*)d_in[13],
             (const float*)d_in[20], (const float*)d_in[21],
             (const float*)d_in[22], (const float*)d_in[23],
             (const float*)d_in[24], (const float*)d_in[25],
             (const float*)d_in[30], (const float*)d_in[31],
             (const float*)d_in[32], (const float*)d_in[33],
             F1, K21, xg1, z1);
}
